// GATInnerLayer_20615843020951
// MI455X (gfx1250) — matrix operand order repaired and checked by reading, not compiled, not run on hardware
//
#include <hip/hip_runtime.h>
#include <hip/hip_bf16.h>

#define IN_DIM  256
#define OUT_DIM 128
#define NODES   20000
#define TILE    16          // nodes per workgroup
#define NTHREADS 256        // 8 waves (wave32)

typedef __attribute__((ext_vector_type(16))) __bf16 v16bf;
typedef __attribute__((ext_vector_type(8)))  __bf16 v8bf;
typedef __attribute__((ext_vector_type(8)))  float  v8f;

#define SCALE 0.0625f       // 1/sqrt(256)

// ---------------------------------------------------------------------------
// Kernel 1: convert f32 weights -> bf16, packed [Wq | Wk | Wv], each 128x256
// row-major (reduction dim contiguous). 192 KB total -> stays L2 resident.
// ---------------------------------------------------------------------------
__global__ void cvt_weights_kernel(const float* __restrict__ Wq,
                                   const float* __restrict__ Wk,
                                   const float* __restrict__ Wv,
                                   __bf16* __restrict__ Wbf) {
    int i = blockIdx.x * blockDim.x + threadIdx.x;
    const int sz = OUT_DIM * IN_DIM;
    if (i < sz) {
        Wbf[i]          = (__bf16)Wq[i];
        Wbf[i + sz]     = (__bf16)Wk[i];
        Wbf[i + 2 * sz] = (__bf16)Wv[i];
    }
}

// ---------------------------------------------------------------------------
// Kernel 2: fused projections (WMMA bf16) + per-node softmax attention.
// One workgroup = 16 nodes. 24 WMMA 16x16 output tiles ([Q|K|V] x 8 col-tiles),
// 3 tiles per wave, K=256 in 8 steps of 32.
// ---------------------------------------------------------------------------

__device__ __forceinline__ v16bf make_v16(v8bf lo, v8bf hi) {
    return __builtin_shufflevector(lo, hi, 0,1,2,3,4,5,6,7,8,9,10,11,12,13,14,15);
}

__launch_bounds__(NTHREADS)
__global__ void gat_fused_kernel(const float*  __restrict__ h,
                                 const __bf16* __restrict__ Wbf,
                                 float*        __restrict__ out) {
    // LDS (gfx1250: up to 320 KB/workgroup). Total here ~101 KB.
    __shared__ __bf16 h_lds[TILE][IN_DIM + 8];          // 8448 B, bf16 A-source
    __shared__ float  p_lds[TILE][3 * OUT_DIM + 4];     // 24832 B, q|k|v per node
    __shared__ float  E[OUT_DIM][OUT_DIM + 1];          // 66048 B, exp matrix (padded)
    __shared__ float  dpart[2][OUT_DIM];                // partial denominators
    __shared__ float  wvec[OUT_DIM];                    // v_j / denom_j
    __shared__ float  qmm[2];                           // {min q, max q}

    const int t     = threadIdx.x;
    const int lane  = t & 31;
    const int wave  = t >> 5;
    const int node0 = blockIdx.x * TILE;

    // ---- stage h tile into LDS as bf16 (coalesced: row p, col t) ----
#pragma unroll
    for (int p = 0; p < TILE; ++p)
        h_lds[p][t] = (__bf16)h[(long)(node0 + p) * IN_DIM + t];
    __syncthreads();

    // ---- WMMA projections ----
    // 16-bit layouts as used here (wave32, contiguous k order):
    //   A 16x32: lane = 16*hi + r holds row r; vec elems 0..15 -> K = 16*hi + 0..15.
    //   B 32x16: lane = 16*hi + r holds col r; vec elems 0..15 -> K = 16*hi + 0..15.
    //   C/D 16x16 f32: vec elem g -> row g + 8*hi, col r.
    const int r  = lane & 15;
    const int hi = lane >> 4;

    const int tt0 = wave * 3;           // tile ids 0..23: (tt>>3)=matrix, (tt&7)=col-tile
    v8f acc0 = {}, acc1 = {}, acc2 = {};

    // per-lane base element offsets into Wbf for each of this wave's 3 tiles
    long b0 = (long)(((tt0 + 0) >> 3) * OUT_DIM + ((tt0 + 0) & 7) * 16 + r) * IN_DIM + hi * 16;
    long b1 = (long)(((tt0 + 1) >> 3) * OUT_DIM + ((tt0 + 1) & 7) * 16 + r) * IN_DIM + hi * 16;
    long b2 = (long)(((tt0 + 2) >> 3) * OUT_DIM + ((tt0 + 2) & 7) * 16 + r) * IN_DIM + hi * 16;

#pragma unroll
    for (int k0 = 0; k0 < IN_DIM; k0 += 32) {
        // A fragment from LDS (shared by all 3 tiles of this wave)
        const __bf16* ap = &h_lds[r][k0 + hi * 16];
        v16bf a = make_v16(*(const v8bf*)ap, *(const v8bf*)(ap + 8));

        const __bf16* p0 = Wbf + b0 + k0;
        const __bf16* p1 = Wbf + b1 + k0;
        const __bf16* p2 = Wbf + b2 + k0;
        v16bf w0 = make_v16(*(const v8bf*)p0, *(const v8bf*)(p0 + 8));
        v16bf w1 = make_v16(*(const v8bf*)p1, *(const v8bf*)(p1 + 8));
        v16bf w2 = make_v16(*(const v8bf*)p2, *(const v8bf*)(p2 + 8));

        acc0 = __builtin_amdgcn_wmma_f32_16x16x32_bf16(false, a, false, w0, (short)0, acc0, false, false);
        acc1 = __builtin_amdgcn_wmma_f32_16x16x32_bf16(false, a, false, w1, (short)0, acc1, false, false);
        acc2 = __builtin_amdgcn_wmma_f32_16x16x32_bf16(false, a, false, w2, (short)0, acc2, false, false);
    }

    // store D tiles to p_lds:  col = matrix*128 + coltile*16 + r, row = g + 8*hi
    {
        const int c0 = ((tt0 + 0) >> 3) * OUT_DIM + ((tt0 + 0) & 7) * 16 + r;
        const int c1 = ((tt0 + 1) >> 3) * OUT_DIM + ((tt0 + 1) & 7) * 16 + r;
        const int c2 = ((tt0 + 2) >> 3) * OUT_DIM + ((tt0 + 2) & 7) * 16 + r;
#pragma unroll
        for (int g = 0; g < 8; ++g) {
            p_lds[g + 8 * hi][c0] = acc0[g];
            p_lds[g + 8 * hi][c1] = acc1[g];
            p_lds[g + 8 * hi][c2] = acc2[g];
        }
    }
    __syncthreads();

    // ---- per-node attention: softmax over i of (q_i * c_j), weighted by v ----
    const int jj   = t & 127;     // column owned in phase 2
    const int half = t >> 7;      // split i-range between thread halves

    for (int n = 0; n < TILE; ++n) {
        // q min/max for a numerically exact softmax-over-i shift per column
        if (wave == 0) {
            float mx = -3.0e38f, mn = 3.0e38f;
#pragma unroll
            for (int p = 0; p < 4; ++p) {
                float qv = p_lds[n][lane + 32 * p];
                mx = fmaxf(mx, qv); mn = fminf(mn, qv);
            }
#pragma unroll
            for (int m = 16; m >= 1; m >>= 1) {
                mx = fmaxf(mx, __shfl_xor(mx, m, 32));
                mn = fminf(mn, __shfl_xor(mn, m, 32));
            }
            if (lane == 0) { qmm[0] = mn; qmm[1] = mx; }
        }
        __syncthreads();

        // E[i][jj] = exp(q_i * c_jj - m_jj); each exp computed exactly once
        {
            const float c  = p_lds[n][OUT_DIM + jj] * SCALE;
            const float mj = c * (c >= 0.0f ? qmm[1] : qmm[0]);
            float dsum = 0.0f;
            const int i0 = half * 64;
#pragma unroll 8
            for (int i = i0; i < i0 + 64; ++i) {
                float e = __expf(p_lds[n][i] * c - mj);
                E[i][jj] = e;
                dsum += e;
            }
            dpart[half][jj] = dsum;
        }
        __syncthreads();

        if (t < OUT_DIM)
            wvec[t] = p_lds[n][2 * OUT_DIM + t] / (dpart[0][t] + dpart[1][t]);
        __syncthreads();

        // out[n][i] = sum_j E[i][j] * w[j]   (row i = t; conflict-free: (129*t+j)%64)
        if (t < OUT_DIM) {
            float acc = 0.0f;
#pragma unroll 8
            for (int j = 0; j < OUT_DIM; ++j)
                acc += E[t][j] * wvec[j];
            out[(long)(node0 + n) * OUT_DIM + t] = acc;
        }
        __syncthreads();   // E reused next node
    }
}

// ---------------------------------------------------------------------------
extern "C" void kernel_launch(void* const* d_in, const int* in_sizes, int n_in,
                              void* d_out, int out_size, void* d_ws, size_t ws_size,
                              hipStream_t stream) {
    const float* h  = (const float*)d_in[0];
    const float* Wq = (const float*)d_in[1];
    const float* Wk = (const float*)d_in[2];
    const float* Wv = (const float*)d_in[3];
    float* out = (float*)d_out;

    __bf16* Wbf = (__bf16*)d_ws;   // 3*128*256*2 = 192 KB scratch

    const int wsz = OUT_DIM * IN_DIM;
    cvt_weights_kernel<<<(wsz + 255) / 256, 256, 0, stream>>>(Wq, Wk, Wv, Wbf);
    gat_fused_kernel<<<NODES / TILE, NTHREADS, 0, stream>>>(h, Wbf, out);
}
